// LinearAttention_53609781789331
// MI455X (gfx1250) — matrix operand order repaired and checked by reading, not compiled, not run on hardware
//
#include <hip/hip_runtime.h>
#include <hip/hip_bf16.h>

// ---------------------------------------------------------------------------
// LinearAttention block for MI455X (gfx1250, wave32, WMMA)
// B=16, C=128, H=W=128 (N=16384), HIDDEN=128, 3*HIDDEN=384, NUM_HEADS=4,
// HEAD_DIM=32, GROUPS=32 (4 ch/group), EPS=1e-5
// ---------------------------------------------------------------------------

#define BATCH   16
#define CCH     128
#define TN      16384          // H*W
#define O3      384            // 3*HIDDEN
#define HID     128
#define NH      4
#define HD      32
#define NGRP    32
#define EPSV    1e-5f

typedef __attribute__((ext_vector_type(16))) __bf16 v16bf;
typedef __attribute__((ext_vector_type(8)))  float  v8f;
typedef __attribute__((ext_vector_type(2)))  float  v2f;

union Frag16 { v16bf v; unsigned int u[8]; };

__device__ __forceinline__ unsigned short f2bf(float f) {
    unsigned u = __float_as_uint(f);
    unsigned r = u + 0x7FFFu + ((u >> 16) & 1u);   // round-to-nearest-even
    return (unsigned short)(r >> 16);
}

// ---------------------------------------------------------------------------
// K1: GroupNorm statistics. One block per (b, group); 4*16384 = 65536
// contiguous floats per group.
// ---------------------------------------------------------------------------
__global__ __launch_bounds__(256) void gn_stats_kernel(
    const float* __restrict__ x, float* __restrict__ mu, float* __restrict__ rsig)
{
    int bg = blockIdx.x;                 // b*32 + g
    int t  = threadIdx.x;
    const float4* p = (const float4*)(x + (size_t)bg * 4 * TN);
    float s = 0.f, sq = 0.f;
    for (int i = t; i < 16384; i += 256) {       // 16384 float4 = 65536 floats
        float4 v = p[i];
        s  += v.x + v.y + v.z + v.w;
        sq += v.x*v.x + v.y*v.y + v.z*v.z + v.w*v.w;
    }
    __shared__ float sA[256], sB[256];
    sA[t] = s; sB[t] = sq; __syncthreads();
    for (int st = 128; st > 0; st >>= 1) {
        if (t < st) { sA[t] += sA[t+st]; sB[t] += sB[t+st]; }
        __syncthreads();
    }
    if (t == 0) {
        float m = sA[0] * (1.0f/65536.0f);
        float v = sB[0] * (1.0f/65536.0f) - m*m;
        mu[bg]   = m;
        rsig[bg] = rsqrtf(v + EPSV);
    }
}

// K1b: per-(b,c) scale/shift from GN affine + stats.
__global__ __launch_bounds__(256) void prep_kernel(
    const float* __restrict__ gn_w, const float* __restrict__ gn_b,
    const float* __restrict__ mu, const float* __restrict__ rsig,
    float* __restrict__ sArr, float* __restrict__ tArr)
{
    int idx = blockIdx.x * 256 + threadIdx.x;    // b*128 + c, 2048 total
    if (idx >= BATCH*CCH) return;
    int b = idx >> 7, c = idx & 127, g = c >> 2;
    float s = gn_w[c] * rsig[b*NGRP + g];
    sArr[idx] = s;
    tArr[idx] = gn_b[c] - mu[b*NGRP + g] * s;
}

// K1c: fold GN shift into per-(b,o) QKV bias: qb[b,o] = sum_c W[o,c]*t[b,c].
__global__ __launch_bounds__(256) void qb_kernel(
    const float* __restrict__ qkv_w, const float* __restrict__ tArr,
    float* __restrict__ qb)
{
    int idx = blockIdx.x * 256 + threadIdx.x;    // b*384 + o
    if (idx >= BATCH*O3) return;
    int b = idx / O3, o = idx % O3;
    const float* w = qkv_w + (size_t)o * CCH;
    const float* tv = tArr + b * CCH;
    float acc = 0.f;
    #pragma unroll 8
    for (int c = 0; c < CCH; c++) acc += w[c] * tv[c];
    qb[idx] = acc;
}

// ---------------------------------------------------------------------------
// K2/K9: shared bf16 WMMA GEMM. Out[b,m,n] = sum_k A'[m,k]*Bsrc[b,k,n] + bias
//   A' = A[m,k] * (ascale ? ascale[b,k] : 1)   (A row-major, K=128 contiguous)
//   + optional per-(b,m) bias, per-m bias, residual[b,m,n]
// Block tile 128(M) x 128(N), K-loop 4 steps of 32, 8 waves (one 16-row strip
// each, 8 N-subtiles). v_wmma_f32_16x16x32_bf16 per subtile per K-step.
// ---------------------------------------------------------------------------
__global__ __launch_bounds__(256) void gemm_bf16_kernel(
    const float* __restrict__ A, const float* __restrict__ ascale,
    const float* __restrict__ bias_bm, const float* __restrict__ bias_m,
    const float* __restrict__ Bsrc, const float* __restrict__ resid,
    float* __restrict__ Out, int M)
{
    __shared__ unsigned short lA[128*36];   // [m_local][k], padded stride 36
    __shared__ unsigned short lB[128*36];   // [n_local][k] (K-contiguous)
    int n0 = blockIdx.x * 128;
    int m0 = blockIdx.y * 128;
    int b  = blockIdx.z;
    int t  = threadIdx.x;
    int wave = t >> 5, lane = t & 31, col = lane & 15;
    bool hi = lane >= 16;
    const float* Bb = Bsrc + (size_t)b * CCH * TN;

    v8f acc[8];
    #pragma unroll
    for (int i = 0; i < 8; i++) acc[i] = (v8f){0.f,0.f,0.f,0.f,0.f,0.f,0.f,0.f};

    for (int ks = 0; ks < 128; ks += 32) {
        { // stage A (128 x 32), apply per-(b,k) scale
            int r  = t >> 1;
            int kh = (t & 1) * 16;
            const float* ap = A + (size_t)(m0 + r) * CCH + ks + kh;
            unsigned short* dst = &lA[r*36 + kh];
            if (ascale) {
                const float* sp = ascale + (size_t)b * CCH + ks + kh;
                #pragma unroll
                for (int j = 0; j < 16; j++) dst[j] = f2bf(ap[j] * sp[j]);
            } else {
                #pragma unroll
                for (int j = 0; j < 16; j++) dst[j] = f2bf(ap[j]);
            }
        }
        { // stage B transposed: lB[n][k], source rows are n-contiguous
            int kr = t >> 3;
            int j0 = (t & 7) * 16;
            const float* bp = Bb + (size_t)(ks + kr) * TN + n0 + j0;
            #pragma unroll
            for (int j = 0; j < 16; j++) lB[(j0 + j)*36 + kr] = f2bf(bp[j]);
        }
        __syncthreads();

        // A fragment: lanes 0-15 row=col K={0..7,16..23}; lanes 16-31 K+8
        Frag16 af;
        {
            const unsigned short* base = &lA[(wave*16 + col)*36];
            int kb = hi ? 8 : 0;
            #pragma unroll
            for (int p = 0; p < 4; p++) {
                af.u[p]   = *(const unsigned int*)(base + kb + 2*p);
                af.u[4+p] = *(const unsigned int*)(base + kb + 16 + 2*p);
            }
        }
        #pragma unroll
        for (int ns = 0; ns < 8; ns++) {
            Frag16 bf_;
            const unsigned short* base = &lB[(ns*16 + col)*36 + (hi ? 8 : 0)];
            #pragma unroll
            for (int p = 0; p < 8; p++) bf_.u[p] = *(const unsigned int*)(base + 2*(p & 3) + ((p >> 2) << 4));
            acc[ns] = __builtin_amdgcn_wmma_f32_16x16x32_bf16(
                false, af.v, false, bf_.v, (short)0, acc[ns], false, false);
        }
        __syncthreads();
    }

    // epilogue: element i of acc[ns] is (m = wave*16 + i + 8*hi, n = ns*16+col)
    #pragma unroll
    for (int ns = 0; ns < 8; ns++) {
        #pragma unroll
        for (int i = 0; i < 8; i++) {
            int m = m0 + wave*16 + i + (hi ? 8 : 0);
            int n = n0 + ns*16 + col;
            float v = acc[ns][i];
            if (bias_bm) v += bias_bm[b*M + m];
            if (bias_m)  v += bias_m[m];
            size_t oidx = ((size_t)b*M + m) * TN + n;
            if (resid)   v += resid[oidx];
            Out[oidx] = v;
        }
    }
}

// ---------------------------------------------------------------------------
// K3: q softmax stats per row (b,o<128): max and sum(exp).
// ---------------------------------------------------------------------------
__global__ __launch_bounds__(256) void qstat_kernel(
    const float* __restrict__ qkv, float* __restrict__ qmax, float* __restrict__ qsum)
{
    int r = blockIdx.x;                     // b*128 + o
    int b = r >> 7, o = r & 127;
    const float* row = qkv + ((size_t)b*O3 + o) * TN;
    int t = threadIdx.x;
    __shared__ float sm[256];
    float m = -3.402823e38f;
    for (int i = t; i < TN; i += 256) m = fmaxf(m, row[i]);
    sm[t] = m; __syncthreads();
    for (int s = 128; s > 0; s >>= 1) { if (t < s) sm[t] = fmaxf(sm[t], sm[t+s]); __syncthreads(); }
    float M = sm[0]; __syncthreads();
    float su = 0.f;
    for (int i = t; i < TN; i += 256) su += __expf(row[i] - M);
    sm[t] = su; __syncthreads();
    for (int s = 128; s > 0; s >>= 1) { if (t < s) sm[t] += sm[t+s]; __syncthreads(); }
    if (t == 0) { qmax[r] = M; qsum[r] = sm[0]; }
}

// K4: k softmax stats per column (b,h,n): max/sum over d (32 regs per thread).
__global__ __launch_bounds__(256) void kstat_kernel(
    const float* __restrict__ qkv, float* __restrict__ kmax, float* __restrict__ ksum)
{
    int idx = blockIdx.x * 256 + threadIdx.x;   // (b*4+h)*16384 + n
    int n  = idx & (TN-1);
    int bh = idx >> 14;
    int b = bh >> 2, h = bh & 3;
    const float* base = qkv + ((size_t)(b*O3 + HID + h*HD)) * TN + n;
    float vals[HD];
    #pragma unroll
    for (int d = 0; d < HD; d++) vals[d] = base[(size_t)d * TN];
    float m = vals[0];
    #pragma unroll
    for (int d = 1; d < HD; d++) m = fmaxf(m, vals[d]);
    float s = 0.f;
    #pragma unroll
    for (int d = 0; d < HD; d++) s += __expf(vals[d] - m);
    kmax[idx] = m; ksum[idx] = s;
}

// ---------------------------------------------------------------------------
// K5: ctx[b,h,d,e] = sum_n k'[d,n] * v[e,n], K=16384, f32 WMMA 16x16x4.
// One block per (b,h); 8 waves split K inside 64-wide staged chunks;
// deterministic LDS cross-wave reduction.
// ---------------------------------------------------------------------------
__global__ __launch_bounds__(256) void ctx_kernel(
    const float* __restrict__ qkv, const float* __restrict__ kmax,
    const float* __restrict__ ksum, float* __restrict__ ctxws)
{
    __shared__ float kt[32*66];
    __shared__ float vt[32*66];
    __shared__ float red[8*1024];
    int bh = blockIdx.x;
    int b = bh >> 2, h = bh & 3;
    int t = threadIdx.x, wave = t >> 5, lane = t & 31, col = lane & 15;
    bool hi = lane >= 16;
    const float* kbase = qkv + (size_t)(b*O3 + HID  + h*HD) * TN;
    const float* vbase = qkv + (size_t)(b*O3 + 2*HID + h*HD) * TN;
    const float* km  = kmax + (size_t)bh * TN;
    const float* ksm = ksum + (size_t)bh * TN;

    v8f acc[2][2];
    #pragma unroll
    for (int i = 0; i < 2; i++)
        #pragma unroll
        for (int j = 0; j < 2; j++) acc[i][j] = (v8f){0.f,0.f,0.f,0.f,0.f,0.f,0.f,0.f};

    for (int c0 = 0; c0 < TN; c0 += 64) {
        { // stage k' (normalized on the fly) and v, 32 x 64 each
            int d = t >> 3, j0 = (t & 7) * 8;
            const float* kp = kbase + (size_t)d * TN + c0 + j0;
            const float* vp = vbase + (size_t)d * TN + c0 + j0;
            #pragma unroll
            for (int j = 0; j < 8; j++) {
                int n = c0 + j0 + j;
                kt[d*66 + j0 + j] = __expf(kp[j] - km[n]) / ksm[n];
                vt[d*66 + j0 + j] = vp[j];
            }
        }
        __syncthreads();
        // 16 K-groups of 4 in this chunk; wave w takes groups {w, w+8}
        for (int g = wave; g < 16; g += 8) {
            int nl = g*4 + (hi ? 2 : 0);
            #pragma unroll
            for (int sm = 0; sm < 2; sm++) {
                v2f a = { kt[(sm*16 + col)*66 + nl], kt[(sm*16 + col)*66 + nl + 1] };
                #pragma unroll
                for (int sn = 0; sn < 2; sn++) {
                    v2f bb = { vt[(sn*16 + col)*66 + nl], vt[(sn*16 + col)*66 + nl + 1] };
                    acc[sm][sn] = __builtin_amdgcn_wmma_f32_16x16x4_f32(
                        false, a, false, bb, (short)0, acc[sm][sn], false, false);
                }
            }
        }
        __syncthreads();
    }
    // cross-wave reduction (deterministic)
    #pragma unroll
    for (int sm = 0; sm < 2; sm++)
        #pragma unroll
        for (int sn = 0; sn < 2; sn++)
            #pragma unroll
            for (int i = 0; i < 8; i++)
                red[wave*1024 + (sm*2+sn)*256 + (i + (hi?8:0))*16 + col] = acc[sm][sn][i];
    __syncthreads();
    for (int e = t; e < 1024; e += 256) {
        int st = e >> 8, pos = e & 255;
        float s = 0.f;
        #pragma unroll
        for (int w = 0; w < 8; w++) s += red[w*1024 + e];
        int d  = (st >> 1)*16 + (pos >> 4);
        int ee = (st & 1)*16 + (pos & 15);
        ctxws[((size_t)bh*HD + d)*HD + ee] = s;
    }
}

// ---------------------------------------------------------------------------
// K6: mid[b,o=h*32+e,n] = sum_d ctx[b,h,d,e] * q'[b,h*32+d,n]  (K=32, bf16 WMMA)
// ---------------------------------------------------------------------------
__global__ __launch_bounds__(256) void mid_kernel(
    const float* __restrict__ qkv, const float* __restrict__ qmax,
    const float* __restrict__ qsum, const float* __restrict__ ctxws,
    float* __restrict__ midws)
{
    __shared__ unsigned short qT[128*132];   // [n_local][o] bf16 (K-contiguous)
    __shared__ unsigned short cT[4*32*36];   // [h][e][d]     bf16
    int n0 = blockIdx.x * 128;
    int b  = blockIdx.y;
    int t = threadIdx.x, wave = t >> 5, lane = t & 31, col = lane & 15;
    bool hi = lane >= 16;

    for (int idx = t; idx < NH*HD*HD; idx += 256) {   // ctx transpose -> [h][e][d]
        int h = idx >> 10, rem = idx & 1023, d = rem >> 5, e = rem & 31;
        cT[(h*HD + e)*36 + d] = f2bf(ctxws[((size_t)(b*NH + h)*HD + d)*HD + e]);
    }
    { // q' tile, normalized on the fly
        int o = t >> 1, j0 = (t & 1) * 64;
        const float* qp = qkv + ((size_t)b*O3 + o) * TN + n0 + j0;
        float mx  = qmax[b*HID + o];
        float inv = 1.0f / qsum[b*HID + o];
        for (int j = 0; j < 64; j++)
            qT[(j0 + j)*132 + o] = f2bf(__expf(qp[j] - mx) * inv);
    }
    __syncthreads();

    int h = wave >> 1, e0 = (wave & 1) * 16;
    Frag16 af;
    {
        const unsigned short* base = &cT[(h*HD + e0 + col)*36];
        int kb = hi ? 8 : 0;
        #pragma unroll
        for (int p = 0; p < 4; p++) {
            af.u[p]   = *(const unsigned int*)(base + kb + 2*p);
            af.u[4+p] = *(const unsigned int*)(base + kb + 16 + 2*p);
        }
    }
    #pragma unroll
    for (int ns = 0; ns < 8; ns++) {
        Frag16 bf_;
        const unsigned short* base = &qT[(ns*16 + col)*132 + h*HD + (hi ? 8 : 0)];
        #pragma unroll
        for (int p = 0; p < 8; p++) bf_.u[p] = *(const unsigned int*)(base + 2*(p & 3) + ((p >> 2) << 4));
        v8f a0 = (v8f){0.f,0.f,0.f,0.f,0.f,0.f,0.f,0.f};
        a0 = __builtin_amdgcn_wmma_f32_16x16x32_bf16(
            false, af.v, false, bf_.v, (short)0, a0, false, false);
        #pragma unroll
        for (int i = 0; i < 8; i++) {
            int o = h*HD + e0 + i + (hi ? 8 : 0);
            int n = n0 + ns*16 + col;
            midws[((size_t)b*HID + o) * TN + n] = a0[i];
        }
    }
}

// ---------------------------------------------------------------------------
// Launch
// ---------------------------------------------------------------------------
extern "C" void kernel_launch(void* const* d_in, const int* in_sizes, int n_in,
                              void* d_out, int out_size, void* d_ws, size_t ws_size,
                              hipStream_t stream) {
    const float* x      = (const float*)d_in[0];
    const float* gn_w   = (const float*)d_in[1];
    const float* gn_b   = (const float*)d_in[2];
    const float* qkv_w  = (const float*)d_in[3];
    const float* proj_w = (const float*)d_in[4];
    const float* proj_b = (const float*)d_in[5];
    float* out = (float*)d_out;

    float* ws = (float*)d_ws;
    // workspace layout (floats)
    float* qkvb  = ws;                                   // 16*384*16384
    float* midb  = qkvb + (size_t)BATCH*O3*TN;           // 16*128*16384
    float* extra = midb + (size_t)BATCH*HID*TN;
    float* mu    = extra;                 // 512
    float* rsig  = mu    + 512;           // 512
    float* sArr  = rsig  + 512;           // 2048
    float* tArr  = sArr  + 2048;          // 2048
    float* qb    = tArr  + 2048;          // 6144
    float* qmaxv = qb    + 6144;          // 2048
    float* qsumv = qmaxv + 2048;          // 2048
    float* kmaxv = qsumv + 2048;          // 1048576
    float* ksumv = kmaxv + (size_t)BATCH*NH*TN;
    float* ctxws = ksumv + (size_t)BATCH*NH*TN;          // 65536

    gn_stats_kernel<<<BATCH*NGRP, 256, 0, stream>>>(x, mu, rsig);
    prep_kernel<<<(BATCH*CCH + 255)/256, 256, 0, stream>>>(gn_w, gn_b, mu, rsig, sArr, tArr);
    qb_kernel<<<(BATCH*O3 + 255)/256, 256, 0, stream>>>(qkv_w, tArr, qb);

    // QKV GEMM: qkv[b,o,n] = sum_c (W[o,c]*s[b,c]) * x[b,c,n] + qb[b,o]
    gemm_bf16_kernel<<<dim3(TN/128, O3/128, BATCH), 256, 0, stream>>>(
        qkv_w, sArr, qb, nullptr, x, nullptr, qkvb, O3);

    qstat_kernel<<<BATCH*HID, 256, 0, stream>>>(qkvb, qmaxv, qsumv);
    kstat_kernel<<<(BATCH*NH*TN)/256, 256, 0, stream>>>(qkvb, kmaxv, ksumv);

    ctx_kernel<<<BATCH*NH, 256, 0, stream>>>(qkvb, kmaxv, ksumv, ctxws);

    mid_kernel<<<dim3(TN/128, BATCH), 256, 0, stream>>>(qkvb, qmaxv, qsumv, ctxws, midb);

    // Projection + bias + residual: out = proj_w @ mid + proj_b + x
    gemm_bf16_kernel<<<dim3(TN/128, HID/128, BATCH), 256, 0, stream>>>(
        proj_w, nullptr, nullptr, proj_b, midb, x, out, HID);
}
